// RankListNetPool_34600256537540
// MI455X (gfx1250) — matrix operand order repaired and checked by reading, not compiled, not run on hardware
//
#include <hip/hip_runtime.h>

#define NVN 100000
#define NCN 100000
#define NEDGE 1600000
#define FDIM 32
#define HDIM 128
#define OUTD 64
#define GNUM 16

typedef __attribute__((ext_vector_type(16))) __bf16 v16bf;
typedef __attribute__((ext_vector_type(16))) unsigned short v16us;
typedef __attribute__((ext_vector_type(8))) float v8f;

__device__ __forceinline__ unsigned short f2bf(float f) {
    unsigned int u = __builtin_bit_cast(unsigned int, f);
    u += 0x7FFFu + ((u >> 16) & 1u);   // round-to-nearest-even to bf16
    return (unsigned short)(u >> 16);
}

// ---------------------------------------------------------------------------
// Pack W[K][N] (row-major fp32) into bf16 WMMA B-fragments.
// Fragment layout (per 32x16 k-tile x n-tile): lane l holds column n = nt*16+(l&15),
// K values ktile*32 + (l>>4)*8 + j for j in 0..7 and ktile*32 + 16 + (l>>4)*8 + (j-8) for j in 8..15.
// ktOff lets two stacked weights share one fragment buffer (concatenated K).
// ---------------------------------------------------------------------------
__global__ void pack_w_kernel(const float* __restrict__ W, int K, int N,
                              unsigned short* __restrict__ dst, int ktOff) {
    int lane = threadIdx.x;                  // 0..31
    int kt = blockIdx.x;                     // k-tile within this matrix
    int nt = blockIdx.y;                     // n-tile
    int ntiles = N >> 4;
    int col = nt * 16 + (lane & 15);
    int kb  = kt * 32 + ((lane >> 4) * 8);
    unsigned short* d = dst + ((((size_t)(ktOff + kt) * ntiles + nt) * 32 + lane) * 16);
#pragma unroll
    for (int j = 0; j < 16; ++j)
        d[j] = f2bf(W[(size_t)(kb + (j < 8 ? j : j + 8)) * N + col]);
    (void)K;
}

// ---------------------------------------------------------------------------
// Edge-parallel scatter-add: each thread handles one edge x 4 features
// (b128 gather + 4 f32 L2 atomics). qshift = log2(Fdim/4).
// ---------------------------------------------------------------------------
__global__ void scatter_add_kernel(const float* __restrict__ xsrc,
                                   const int* __restrict__ src,
                                   const int* __restrict__ dst,
                                   long long nedges, int qshift,
                                   float* __restrict__ accum,
                                   float* __restrict__ cnt) {
    long long total  = nedges << qshift;
    long long stride = (long long)gridDim.x * blockDim.x;
    int qmask = (1 << qshift) - 1;
    int fsh = qshift + 2;                    // log2(Fdim)
    for (long long i = (long long)blockIdx.x * blockDim.x + threadIdx.x;
         i < total; i += stride) {
        long long e = i >> qshift;
        int q = (int)(i & qmask);
        int f = q << 2;
        int s = src[e];
        int d = dst[e];
        float4 v = *(const float4*)(xsrc + (((size_t)s << fsh) + f));
        float* a = accum + (((size_t)d << fsh) + f);
        atomicAdd(a + 0, v.x);
        atomicAdd(a + 1, v.y);
        atomicAdd(a + 2, v.z);
        atomicAdd(a + 3, v.w);
        if (cnt && q == 0) atomicAdd(cnt + d, 1.0f);
    }
}

// ---------------------------------------------------------------------------
// Fused dual-input WMMA GEMM (bf16 inputs, f32 accumulate), K fully unrolled:
//   out = act( (A1 .* 1/max(deg,1)) @ W1  +  A2 @ W2  + bias )
// W1|W2 pre-packed (concatenated K) in Wfrag. One wave = one 16x16 tile.
// A-fragment per ISA: lanes 0-15 -> K 0..7 & 16..23 ; lanes 16-31 -> K 8..15 & 24..31.
// ---------------------------------------------------------------------------
template <int KT1, int KT2>
__global__ void __launch_bounds__(128)
gemm_wmma_kernel(const float* __restrict__ A1,
                 const float* __restrict__ rscale,
                 const float* __restrict__ A2,
                 const unsigned short* __restrict__ Wfrag,
                 const float* __restrict__ bias,
                 float* __restrict__ out, int N, int doRelu) {
    constexpr int K1 = KT1 * 32;
    constexpr int K2 = KT2 * 32;
    int lane = threadIdx.x;                          // 0..31 (one wave per y-row)
    int tm = blockIdx.x;
    int tn = blockIdx.y * blockDim.y + threadIdx.y;  // n-tile
    int ntiles = N >> 4;

    int row = tm * 16 + (lane & 15);      // A row this lane feeds
    int col = tn * 16 + (lane & 15);      // C/D column this lane owns
    int ks0 = (lane >> 4) * 8;            // A k-chunk base within a 32-k tile

    float rs = 1.0f;
    if (rscale) rs = 1.0f / fmaxf(rscale[row], 1.0f);

    v8f c = {};

#pragma unroll
    for (int kt = 0; kt < KT1; ++kt) {
        const float* ap = A1 + (size_t)row * K1 + kt * 32 + ks0;
        float4 p0 = *(const float4*)(ap + 0);
        float4 p1 = *(const float4*)(ap + 4);
        float4 p2 = *(const float4*)(ap + 16);
        float4 p3 = *(const float4*)(ap + 20);
        v16us au;
        au[0]  = f2bf(p0.x * rs); au[1]  = f2bf(p0.y * rs);
        au[2]  = f2bf(p0.z * rs); au[3]  = f2bf(p0.w * rs);
        au[4]  = f2bf(p1.x * rs); au[5]  = f2bf(p1.y * rs);
        au[6]  = f2bf(p1.z * rs); au[7]  = f2bf(p1.w * rs);
        au[8]  = f2bf(p2.x * rs); au[9]  = f2bf(p2.y * rs);
        au[10] = f2bf(p2.z * rs); au[11] = f2bf(p2.w * rs);
        au[12] = f2bf(p3.x * rs); au[13] = f2bf(p3.y * rs);
        au[14] = f2bf(p3.z * rs); au[15] = f2bf(p3.w * rs);
        v16bf a = __builtin_bit_cast(v16bf, au);
        const v16us* bp = (const v16us*)(Wfrag +
            ((((size_t)kt * ntiles + tn) * 32 + lane) * 16));
        v16bf b = __builtin_bit_cast(v16bf, *bp);
        c = __builtin_amdgcn_wmma_f32_16x16x32_bf16(
                false, a, false, b, (short)0, c, false, false);
    }
#pragma unroll
    for (int kt = 0; kt < KT2; ++kt) {
        const float* ap = A2 + (size_t)row * K2 + kt * 32 + ks0;
        float4 p0 = *(const float4*)(ap + 0);
        float4 p1 = *(const float4*)(ap + 4);
        float4 p2 = *(const float4*)(ap + 16);
        float4 p3 = *(const float4*)(ap + 20);
        v16us au;
        au[0]  = f2bf(p0.x); au[1]  = f2bf(p0.y);
        au[2]  = f2bf(p0.z); au[3]  = f2bf(p0.w);
        au[4]  = f2bf(p1.x); au[5]  = f2bf(p1.y);
        au[6]  = f2bf(p1.z); au[7]  = f2bf(p1.w);
        au[8]  = f2bf(p2.x); au[9]  = f2bf(p2.y);
        au[10] = f2bf(p2.z); au[11] = f2bf(p2.w);
        au[12] = f2bf(p3.x); au[13] = f2bf(p3.y);
        au[14] = f2bf(p3.z); au[15] = f2bf(p3.w);
        v16bf a = __builtin_bit_cast(v16bf, au);
        const v16us* bp = (const v16us*)(Wfrag +
            ((((size_t)(KT1 + kt) * ntiles + tn) * 32 + lane) * 16));
        v16bf b = __builtin_bit_cast(v16bf, *bp);
        c = __builtin_amdgcn_wmma_f32_16x16x32_bf16(
                false, a, false, b, (short)0, c, false, false);
    }

    float bv = bias ? bias[col] : 0.0f;
    int mh = (lane >> 4) * 8;             // C/D: vgpr j -> row j + 8*(lane>=16)
#pragma unroll
    for (int j = 0; j < 8; ++j) {
        float v = c[j] + bv;
        if (doRelu) v = fmaxf(v, 0.0f);
        out[(size_t)(tm * 16 + mh + j) * N + col] = v;
    }
}

// ---------------------------------------------------------------------------
// Segment-sum pooling (per-graph) with count; 4 features per thread.
// ---------------------------------------------------------------------------
__global__ void pool_sum_kernel(const float* __restrict__ z,
                                const int* __restrict__ batch,
                                long long nn, int qshift,
                                float* __restrict__ gsum,
                                float* __restrict__ gcnt) {
    long long total  = nn << qshift;
    long long stride = (long long)gridDim.x * blockDim.x;
    int qmask = (1 << qshift) - 1;
    int fsh = qshift + 2;
    for (long long i = (long long)blockIdx.x * blockDim.x + threadIdx.x;
         i < total; i += stride) {
        long long node = i >> qshift;
        int q = (int)(i & qmask);
        int f = q << 2;
        int b = batch[node];
        float4 v = *(const float4*)(z + (((size_t)node << fsh) + f));
        float* a = gsum + (((size_t)b << fsh) + f);
        atomicAdd(a + 0, v.x);
        atomicAdd(a + 1, v.y);
        atomicAdd(a + 2, v.z);
        atomicAdd(a + 3, v.w);
        if (q == 0) atomicAdd(gcnt + b, 1.0f);
    }
}

__global__ void pool_final_kernel(const float* __restrict__ gsumV,
                                  const float* __restrict__ gcntV,
                                  const float* __restrict__ gsumC,
                                  const float* __restrict__ gcntC,
                                  float* __restrict__ gmat) {
    int i = blockIdx.x * blockDim.x + threadIdx.x;
    if (i >= GNUM * 2 * OUTD) return;
    int g = i >> 7;            // 128 features per graph row
    int f = i & 127;
    float v = (f < OUTD) ? gsumV[g * OUTD + f] / fmaxf(gcntV[g], 1.0f)
                         : gsumC[g * OUTD + (f - OUTD)] / fmaxf(gcntC[g], 1.0f);
    gmat[i] = v;
}

// Final per-task heads: out[g,t] = dot(th_t[g,:], W1_t) + b1_t
__global__ void heads_kernel(const float* __restrict__ th0,
                             const float* __restrict__ th1,
                             const float* __restrict__ w10, const float* __restrict__ b10,
                             const float* __restrict__ w11, const float* __restrict__ b11,
                             float* __restrict__ out) {
    int tid = threadIdx.x;
    if (tid >= 32) return;
    int t = tid >> 4;
    int g = tid & 15;
    const float* h = t ? th1 : th0;
    const float* w = t ? w11 : w10;
    float s = (t ? b11 : b10)[0];
#pragma unroll
    for (int k = 0; k < 64; ++k) s += h[g * 64 + k] * w[k];
    out[g * 2 + t] = s;
}

// ---------------------------------------------------------------------------
extern "C" void kernel_launch(void* const* d_in, const int* in_sizes, int n_in,
                              void* d_out, int out_size, void* d_ws, size_t ws_size,
                              hipStream_t stream) {
    (void)in_sizes; (void)n_in; (void)out_size; (void)ws_size;

    const float* x_vars = (const float*)d_in[0];
    const float* x_cons = (const float*)d_in[1];
    const int* e_cv_src = (const int*)d_in[2];
    const int* e_cv_dst = (const int*)d_in[3];
    const int* e_vc_src = (const int*)d_in[4];
    const int* e_vc_dst = (const int*)d_in[5];
    const int* batch_v  = (const int*)d_in[6];
    const int* batch_c  = (const int*)d_in[7];
    // params in setup_inputs() dict insertion order
    const float* l1_cv_Wl = (const float*)d_in[8];
    const float* l1_cv_Wr = (const float*)d_in[9];
    const float* l1_cv_b  = (const float*)d_in[10];
    const float* l1_vc_Wl = (const float*)d_in[11];
    const float* l1_vc_Wr = (const float*)d_in[12];
    const float* l1_vc_b  = (const float*)d_in[13];
    const float* l2_cv_Wl = (const float*)d_in[14];
    const float* l2_cv_Wr = (const float*)d_in[15];
    const float* l2_cv_b  = (const float*)d_in[16];
    const float* l2_vc_Wl = (const float*)d_in[17];
    const float* l2_vc_Wr = (const float*)d_in[18];
    const float* l2_vc_b  = (const float*)d_in[19];
    const float* jk_v_W   = (const float*)d_in[20];
    const float* jk_v_b   = (const float*)d_in[21];
    const float* jk_c_W   = (const float*)d_in[22];
    const float* jk_c_b   = (const float*)d_in[23];
    const float* pool_W0  = (const float*)d_in[24];
    const float* pool_b0  = (const float*)d_in[25];
    const float* pool_W1  = (const float*)d_in[26];
    const float* pool_b1  = (const float*)d_in[27];
    const float* pool_W2  = (const float*)d_in[28];
    const float* pool_b2  = (const float*)d_in[29];
    const float* h0_W0 = (const float*)d_in[30];
    const float* h0_b0 = (const float*)d_in[31];
    const float* h0_W1 = (const float*)d_in[32];
    const float* h0_b1 = (const float*)d_in[33];
    const float* h1_W0 = (const float*)d_in[34];
    const float* h1_b0 = (const float*)d_in[35];
    const float* h1_W1 = (const float*)d_in[36];
    const float* h1_b1 = (const float*)d_in[37];

    // ---- carve workspace (floats) ----
    float* ws = (float*)d_ws;
    size_t off = 0;
    auto allocF = [&](size_t n) { float* p = ws + off; off += n; return p; };

    // zero-group 1: layer1 accumulators + degrees
    float* m_v   = allocF((size_t)NVN * FDIM);
    float* m_c   = allocF((size_t)NCN * FDIM);
    float* cnt_v = allocF(NVN);
    float* cnt_c = allocF(NCN);
    size_t z1_floats = (size_t)NVN * FDIM + (size_t)NCN * FDIM + NVN + NCN;
    // zero-group 2: layer2 accumulators
    float* agg_v = allocF((size_t)NVN * HDIM);
    float* agg_c = allocF((size_t)NCN * HDIM);
    size_t z2_floats = (size_t)(NVN + NCN) * HDIM;
    // zero-group 3: pooling / MLP small buffers
    float* gsumV = allocF(GNUM * OUTD);
    float* gsumC = allocF(GNUM * OUTD);
    float* gcntV = allocF(GNUM);
    float* gcntC = allocF(GNUM);
    float* gmat  = allocF(GNUM * 2 * OUTD);   // [16,128]
    float* mh0   = allocF(GNUM * 256);
    float* mh1   = allocF(GNUM * 256);
    float* mh2   = allocF(GNUM * 128);
    float* th0   = allocF(GNUM * 64);
    float* th1   = allocF(GNUM * 64);
    size_t z3_floats = (size_t)(GNUM * OUTD * 2 + 2 * GNUM + GNUM * 128 +
                                GNUM * 256 * 2 + GNUM * 128 + GNUM * 64 * 2);
    // big activation buffers
    float* hv  = allocF((size_t)NVN * HDIM);
    float* hc  = allocF((size_t)NCN * HDIM);
    float* hv2 = allocF((size_t)NVN * HDIM);
    float* hc2 = allocF((size_t)NCN * HDIM);
    float* zv = agg_v;   // alias: agg buffers dead after layer2 GEMM
    float* zc = agg_c;

    // bf16 fragment region (32B aligned)
    off = (off + 15) & ~(size_t)15;
    unsigned short* ub = (unsigned short*)(ws + off);
    size_t uoff = 0;
    auto allocU = [&](size_t n) { unsigned short* p = ub + uoff; uoff += n; return p; };
    unsigned short* frag_l1v = allocU(2 * 8 * 512);   // K=64  N=128
    unsigned short* frag_l1c = allocU(2 * 8 * 512);
    unsigned short* frag_l2v = allocU(8 * 8 * 512);   // K=256 N=128
    unsigned short* frag_l2c = allocU(8 * 8 * 512);
    unsigned short* frag_jkv = allocU(4 * 4 * 512);   // K=128 N=64
    unsigned short* frag_jkc = allocU(4 * 4 * 512);
    unsigned short* frag_p0  = allocU(4 * 16 * 512);  // K=128 N=256
    unsigned short* frag_p1  = allocU(8 * 16 * 512);  // K=256 N=256
    unsigned short* frag_p2  = allocU(8 * 8 * 512);   // K=256 N=128
    unsigned short* frag_h0  = allocU(4 * 4 * 512);   // K=128 N=64
    unsigned short* frag_h1  = allocU(4 * 4 * 512);

    // ---- zero accumulators ----
    hipMemsetAsync(m_v,   0, z1_floats * sizeof(float), stream);
    hipMemsetAsync(agg_v, 0, z2_floats * sizeof(float), stream);
    hipMemsetAsync(gsumV, 0, z3_floats * sizeof(float), stream);

    // ---- pack weights to bf16 fragments ----
    dim3 pb(32);
    pack_w_kernel<<<dim3(1, 8),  pb, 0, stream>>>(l1_cv_Wl, 32, 128, frag_l1v, 0);
    pack_w_kernel<<<dim3(1, 8),  pb, 0, stream>>>(l1_cv_Wr, 32, 128, frag_l1v, 1);
    pack_w_kernel<<<dim3(1, 8),  pb, 0, stream>>>(l1_vc_Wl, 32, 128, frag_l1c, 0);
    pack_w_kernel<<<dim3(1, 8),  pb, 0, stream>>>(l1_vc_Wr, 32, 128, frag_l1c, 1);
    pack_w_kernel<<<dim3(4, 8),  pb, 0, stream>>>(l2_cv_Wl, 128, 128, frag_l2v, 0);
    pack_w_kernel<<<dim3(4, 8),  pb, 0, stream>>>(l2_cv_Wr, 128, 128, frag_l2v, 4);
    pack_w_kernel<<<dim3(4, 8),  pb, 0, stream>>>(l2_vc_Wl, 128, 128, frag_l2c, 0);
    pack_w_kernel<<<dim3(4, 8),  pb, 0, stream>>>(l2_vc_Wr, 128, 128, frag_l2c, 4);
    pack_w_kernel<<<dim3(4, 4),  pb, 0, stream>>>(jk_v_W, 128, 64, frag_jkv, 0);
    pack_w_kernel<<<dim3(4, 4),  pb, 0, stream>>>(jk_c_W, 128, 64, frag_jkc, 0);
    pack_w_kernel<<<dim3(4, 16), pb, 0, stream>>>(pool_W0, 128, 256, frag_p0, 0);
    pack_w_kernel<<<dim3(8, 16), pb, 0, stream>>>(pool_W1, 256, 256, frag_p1, 0);
    pack_w_kernel<<<dim3(8, 8),  pb, 0, stream>>>(pool_W2, 256, 128, frag_p2, 0);
    pack_w_kernel<<<dim3(4, 4),  pb, 0, stream>>>(h0_W0, 128, 64, frag_h0, 0);
    pack_w_kernel<<<dim3(4, 4),  pb, 0, stream>>>(h1_W0, 128, 64, frag_h1, 0);

    // ---- layer 1: scatter-mean (F=32) then fused dual GEMM + relu ----
    scatter_add_kernel<<<16384, 256, 0, stream>>>(x_cons, e_cv_src, e_cv_dst,
                                                  (long long)NEDGE, 3, m_v, cnt_v);
    scatter_add_kernel<<<16384, 256, 0, stream>>>(x_vars, e_vc_src, e_vc_dst,
                                                  (long long)NEDGE, 3, m_c, cnt_c);
    dim3 gblk(32, 4);
    gemm_wmma_kernel<1, 1><<<dim3(NVN / 16, 2), gblk, 0, stream>>>(
        m_v, cnt_v, x_vars, frag_l1v, l1_cv_b, hv, 128, 1);
    gemm_wmma_kernel<1, 1><<<dim3(NCN / 16, 2), gblk, 0, stream>>>(
        m_c, cnt_c, x_cons, frag_l1c, l1_vc_b, hc, 128, 1);

    // ---- layer 2: scatter-mean (H=128, same edge degrees) + dual GEMM ----
    scatter_add_kernel<<<16384, 256, 0, stream>>>(hc, e_cv_src, e_cv_dst,
                                                  (long long)NEDGE, 5, agg_v, nullptr);
    scatter_add_kernel<<<16384, 256, 0, stream>>>(hv, e_vc_src, e_vc_dst,
                                                  (long long)NEDGE, 5, agg_c, nullptr);
    gemm_wmma_kernel<4, 4><<<dim3(NVN / 16, 2), gblk, 0, stream>>>(
        agg_v, cnt_v, hv, frag_l2v, l2_cv_b, hv2, 128, 1);
    gemm_wmma_kernel<4, 4><<<dim3(NCN / 16, 2), gblk, 0, stream>>>(
        agg_c, cnt_c, hc, frag_l2c, l2_vc_b, hc2, 128, 1);

    // ---- JK linear (no relu) ----
    gemm_wmma_kernel<4, 0><<<dim3(NVN / 16, 1), gblk, 0, stream>>>(
        hv2, nullptr, nullptr, frag_jkv, jk_v_b, zv, 64, 0);
    gemm_wmma_kernel<4, 0><<<dim3(NCN / 16, 1), gblk, 0, stream>>>(
        hc2, nullptr, nullptr, frag_jkc, jk_c_b, zc, 64, 0);

    // ---- global mean pool per graph, concat ----
    pool_sum_kernel<<<8192, 256, 0, stream>>>(zv, batch_v, (long long)NVN, 4, gsumV, gcntV);
    pool_sum_kernel<<<8192, 256, 0, stream>>>(zc, batch_c, (long long)NCN, 4, gsumC, gcntC);
    pool_final_kernel<<<8, 256, 0, stream>>>(gsumV, gcntV, gsumC, gcntC, gmat);

    // ---- pooling MLP + task-head hidden layers (M=16, one WMMA tile row) ----
    gemm_wmma_kernel<4, 0><<<dim3(1, 4), gblk, 0, stream>>>(
        gmat, nullptr, nullptr, frag_p0, pool_b0, mh0, 256, 1);
    gemm_wmma_kernel<8, 0><<<dim3(1, 4), gblk, 0, stream>>>(
        mh0, nullptr, nullptr, frag_p1, pool_b1, mh1, 256, 1);
    gemm_wmma_kernel<8, 0><<<dim3(1, 2), gblk, 0, stream>>>(
        mh1, nullptr, nullptr, frag_p2, pool_b2, mh2, 128, 0);
    gemm_wmma_kernel<4, 0><<<dim3(1, 1), gblk, 0, stream>>>(
        mh2, nullptr, nullptr, frag_h0, h0_b0, th0, 64, 1);
    gemm_wmma_kernel<4, 0><<<dim3(1, 1), gblk, 0, stream>>>(
        mh2, nullptr, nullptr, frag_h1, h1_b0, th1, 64, 1);

    heads_kernel<<<1, 32, 0, stream>>>(th0, th1, h0_W1, h0_b1, h1_W1, h1_b1,
                                       (float*)d_out);
}
